// SelfAttentionBlock_86638080295724
// MI455X (gfx1250) — matrix operand order repaired and checked by reading, not compiled, not run on hardware
//
#include <hip/hip_runtime.h>
#include <cstdint>
#include <cstddef>

typedef __bf16 bf16;
typedef __attribute__((ext_vector_type(16))) __bf16 v16bf;
typedef __attribute__((ext_vector_type(8)))  __bf16 v8bf;
typedef __attribute__((ext_vector_type(8)))  float   v8f;

#define BATCH 8
#define SEQ   2048
#define DK    1024   // model dim == output dim

union AFrag { v16bf v; v8bf h[2]; };

static __device__ inline v8f wmma_bf16(v16bf a, v16bf b, v8f c) {
    return __builtin_amdgcn_wmma_f32_16x16x32_bf16(false, a, false, b,
                                                   (short)0, c, false, false);
}

// Load a 16x32 bf16 A fragment (contiguous k = 16*half + s, matching B) for this lane.
// rowptr must already point at (row m for this lane) * stride, k-chunk base.
static __device__ inline v16bf load_a(const bf16* rowptr, unsigned half) {
    AFrag a;
    a.h[0] = *(const v8bf*)(rowptr + half * 16);
    a.h[1] = *(const v8bf*)(rowptr + half * 16 + 8);
    return a.v;
}

// ---------------------------------------------------------------------------
// Kernel 1a: fp32 -> bf16 elementwise convert (x)
// ---------------------------------------------------------------------------
__global__ __launch_bounds__(256) void cvt_f32_bf16(const float* __restrict__ in,
                                                    bf16* __restrict__ out, size_t n) {
    size_t i = (size_t)blockIdx.x * 256 + threadIdx.x;
    if (i < n) out[i] = (bf16)in[i];
}

// ---------------------------------------------------------------------------
// Kernel 1b: transpose W [K][N] f32 -> WT [N][K] bf16 (32x32 LDS tiles)
// ---------------------------------------------------------------------------
__global__ __launch_bounds__(256) void transpose_w(const float* __restrict__ W,
                                                   bf16* __restrict__ WT) {
    __shared__ float t[32][33];
    int n0 = blockIdx.x * 32, k0 = blockIdx.y * 32;
    int tx = threadIdx.x & 31, ty = threadIdx.x >> 5;
    for (int r = ty; r < 32; r += 8)
        t[r][tx] = W[(size_t)(k0 + r) * DK + n0 + tx];
    __syncthreads();
    for (int r = ty; r < 32; r += 8)
        WT[(size_t)(n0 + r) * DK + k0 + tx] = (bf16)t[tx][r];
}

// ---------------------------------------------------------------------------
// Kernel 2: QKV projection GEMM via WMMA bf16, 32x32 tile per wave (2x2 blocking).
//   blockIdx.z: 0=Q (scaled by 1/sqrt(O)), 1=K, 2=V (stored transposed [b][o][s])
// ---------------------------------------------------------------------------
__global__ __launch_bounds__(256) void qkv_gemm(const bf16* __restrict__ X,
                                                const bf16* __restrict__ WTq,
                                                const bf16* __restrict__ WTk,
                                                const bf16* __restrict__ WTv,
                                                const float* __restrict__ bq,
                                                const float* __restrict__ bk,
                                                const float* __restrict__ bv,
                                                bf16* __restrict__ Q,
                                                bf16* __restrict__ K,
                                                bf16* __restrict__ Vt) {
    int which = blockIdx.z;
    const bf16*  WT   = (which == 0) ? WTq : (which == 1) ? WTk : WTv;
    const float* bias = (which == 0) ? bq  : (which == 1) ? bk  : bv;

    int w = threadIdx.x >> 5;
    unsigned lane = threadIdx.x & 31u;
    unsigned nm = lane & 15u, half = lane >> 4;
    int m0 = (blockIdx.y * 8 + w) * 32;                 // 32 rows per wave
    int n0 = blockIdx.x * 32;                           // 32 cols per wave

    const bf16* arow0 = X  + (size_t)(m0 + nm) * DK;
    const bf16* arow1 = X  + (size_t)(m0 + 16 + nm) * DK;
    const bf16* brow0 = WT + (size_t)(n0 + nm) * DK + half * 16;
    const bf16* brow1 = WT + (size_t)(n0 + 16 + nm) * DK + half * 16;

    v8f acc00 = {}, acc01 = {}, acc10 = {}, acc11 = {};
#pragma unroll 2
    for (int kk = 0; kk < DK / 32; ++kk) {
        v16bf a0 = load_a(arow0 + kk * 32, half);
        v16bf a1 = load_a(arow1 + kk * 32, half);
        v16bf b0 = *(const v16bf*)(brow0 + kk * 32);
        v16bf b1 = *(const v16bf*)(brow1 + kk * 32);
        acc00 = wmma_bf16(a0, b0, acc00);
        acc01 = wmma_bf16(a0, b1, acc01);
        acc10 = wmma_bf16(a1, b0, acc10);
        acc11 = wmma_bf16(a1, b1, acc11);
    }

    float bn0 = bias[n0 + nm];
    float bn1 = bias[n0 + 16 + nm];
    if (which < 2) {
        bf16* out = (which == 0) ? Q : K;
        float scale = (which == 0) ? 0.03125f : 1.0f;   // 1/sqrt(1024) folded into Q
#pragma unroll
        for (int r = 0; r < 8; ++r) {
            int M = r + half * 8;
            out[(size_t)(m0 + M) * DK + n0 + nm]           = (bf16)((acc00[r] + bn0) * scale);
            out[(size_t)(m0 + M) * DK + n0 + 16 + nm]      = (bf16)((acc01[r] + bn1) * scale);
            out[(size_t)(m0 + 16 + M) * DK + n0 + nm]      = (bf16)((acc10[r] + bn0) * scale);
            out[(size_t)(m0 + 16 + M) * DK + n0 + 16 + nm] = (bf16)((acc11[r] + bn1) * scale);
        }
    } else {
        int bidx = m0 / SEQ, sloc = m0 % SEQ;           // store V transposed: Vt[b][o][s]
#pragma unroll
        for (int r = 0; r < 8; ++r) {
            int M = r + half * 8;
            Vt[((size_t)bidx * DK + n0 + nm) * SEQ + sloc + M]           = (bf16)(acc00[r] + bn0);
            Vt[((size_t)bidx * DK + n0 + 16 + nm) * SEQ + sloc + M]      = (bf16)(acc01[r] + bn1);
            Vt[((size_t)bidx * DK + n0 + nm) * SEQ + sloc + 16 + M]      = (bf16)(acc10[r] + bn0);
            Vt[((size_t)bidx * DK + n0 + 16 + nm) * SEQ + sloc + 16 + M] = (bf16)(acc11[r] + bn1);
        }
    }
}

// ---------------------------------------------------------------------------
// Kernel 3: causal score tiles S = Q*K^T (Q pre-scaled), 32x32 per wave.
// Only 32-blocks with jt32 <= it32 are computed; diagonal blocks masked.
// ---------------------------------------------------------------------------
__global__ __launch_bounds__(256) void scores_kernel(const bf16* __restrict__ Q,
                                                     const bf16* __restrict__ Km,
                                                     float* __restrict__ SC) {
    int b = blockIdx.z, it32 = blockIdx.x, w = threadIdx.x >> 5;
    int jt32 = blockIdx.y * 8 + w;
    if (jt32 > it32) return;                           // causal: skip upper triangle
    unsigned lane = threadIdx.x & 31u;
    unsigned nm = lane & 15u, half = lane >> 4;
    int i0 = it32 * 32, j0 = jt32 * 32;

    const bf16* qrow0 = Q  + ((size_t)b * SEQ + i0 + nm) * DK;
    const bf16* qrow1 = Q  + ((size_t)b * SEQ + i0 + 16 + nm) * DK;
    const bf16* krow0 = Km + ((size_t)b * SEQ + j0 + nm) * DK + half * 16;
    const bf16* krow1 = Km + ((size_t)b * SEQ + j0 + 16 + nm) * DK + half * 16;

    v8f acc00 = {}, acc01 = {}, acc10 = {}, acc11 = {};
#pragma unroll 2
    for (int kk = 0; kk < DK / 32; ++kk) {
        v16bf a0 = load_a(qrow0 + kk * 32, half);
        v16bf a1 = load_a(qrow1 + kk * 32, half);
        v16bf b0 = *(const v16bf*)(krow0 + kk * 32);
        v16bf b1 = *(const v16bf*)(krow1 + kk * 32);
        acc00 = wmma_bf16(a0, b0, acc00);
        acc01 = wmma_bf16(a0, b1, acc01);
        acc10 = wmma_bf16(a1, b0, acc10);
        acc11 = wmma_bf16(a1, b1, acc11);
    }

    bool diag = (it32 == jt32);
    float* srow = SC + ((size_t)b * SEQ + i0) * SEQ + j0;
#pragma unroll
    for (int r = 0; r < 8; ++r) {
        int M = r + half * 8;
        float v00 = acc00[r], v01 = acc01[r], v10 = acc10[r], v11 = acc11[r];
        if (diag) {                                    // mask j > i within the block
            if ((int)nm > M)           v00 = -1e30f;   // sub (0,0): j-16*0+N vs i-16*0+M
            /* sub (0,1): j = 16+N > i = M always */   v01 = diag ? -1e30f : v01;
            if ((int)nm + 16 > M + 16 && (int)nm > M)  v11 = -1e30f;  // same as N > M
            /* sub (1,0): j = N <= 15 < 16+M = i -> never masked */
        }
        srow[(size_t)M * SEQ + nm]                  = v00;
        srow[(size_t)M * SEQ + 16 + nm]             = v01;
        srow[(size_t)(16 + M) * SEQ + nm]           = v10;
        srow[(size_t)(16 + M) * SEQ + 16 + nm]      = v11;
    }
}

// ---------------------------------------------------------------------------
// Kernel 4: row softmax; writes probabilities as bf16 IN PLACE over f32 scores.
// ---------------------------------------------------------------------------
__global__ __launch_bounds__(256) void softmax_rows(float* __restrict__ SC) {
    int row  = blockIdx.x;                  // b*SEQ + i
    int iloc = row & (SEQ - 1);
    int valid = iloc + 1;
    float* srow = SC + (size_t)row * SEQ;
    bf16*  prow = (bf16*)srow;              // bf16 row aliases f32 row start

    __shared__ float red[256];
    float vals[8], e[8];
    float mx = -3.0e38f;
#pragma unroll
    for (int c = 0; c < 8; ++c) {
        int j = threadIdx.x + c * 256;
        float v = (j < valid) ? srow[j] : -3.0e38f;
        vals[c] = v;
        mx = fmaxf(mx, v);
    }
    red[threadIdx.x] = mx; __syncthreads();
    for (int s = 128; s > 0; s >>= 1) {
        if ((int)threadIdx.x < s) red[threadIdx.x] = fmaxf(red[threadIdx.x], red[threadIdx.x + s]);
        __syncthreads();
    }
    mx = red[0]; __syncthreads();

    float sum = 0.f;
#pragma unroll
    for (int c = 0; c < 8; ++c) {
        int j = threadIdx.x + c * 256;
        e[c] = (j < valid) ? __expf(vals[c] - mx) : 0.f;
        sum += e[c];
    }
    red[threadIdx.x] = sum; __syncthreads();
    for (int s = 128; s > 0; s >>= 1) {
        if ((int)threadIdx.x < s) red[threadIdx.x] += red[threadIdx.x + s];
        __syncthreads();
    }
    float inv = 1.0f / red[0];
    __syncthreads();
#pragma unroll
    for (int c = 0; c < 8; ++c) {
        int j = threadIdx.x + c * 256;
        prow[j] = (bf16)(e[c] * inv);       // zeros for j > i
    }
}

// ---------------------------------------------------------------------------
// Kernel 5: O = P @ V via WMMA, 32x32 per wave. P is bf16 rows embedded in
// the f32 score buffer (row pitch = SEQ floats = 2*SEQ bf16 slots).
// Causal trims the K-loop to it32+1 chunks of 32 keys.
// ---------------------------------------------------------------------------
__global__ __launch_bounds__(256) void av_gemm(const float* __restrict__ SC,
                                               const bf16* __restrict__ Vt,
                                               float* __restrict__ O) {
    int b = blockIdx.z, dt32 = blockIdx.x, w = threadIdx.x >> 5;
    int it32 = blockIdx.y * 8 + w;
    unsigned lane = threadIdx.x & 31u;
    unsigned nm = lane & 15u, half = lane >> 4;
    int i0 = it32 * 32, d0 = dt32 * 32;

    const bf16* pbase = (const bf16*)SC;
    const size_t ppitch = (size_t)SEQ * 2;  // bf16 slots per f32 row
    const bf16* prow0 = pbase + ((size_t)b * SEQ + i0 + nm) * ppitch;
    const bf16* prow1 = pbase + ((size_t)b * SEQ + i0 + 16 + nm) * ppitch;
    const bf16* vrow0 = Vt + ((size_t)b * DK + d0 + nm) * SEQ + half * 16;
    const bf16* vrow1 = Vt + ((size_t)b * DK + d0 + 16 + nm) * SEQ + half * 16;

    int nk = it32 + 1;                      // chunks of 32 keys covering j <= i0+31
    v8f acc00 = {}, acc01 = {}, acc10 = {}, acc11 = {};
    for (int kk = 0; kk < nk; ++kk) {
        v16bf a0 = load_a(prow0 + kk * 32, half);
        v16bf a1 = load_a(prow1 + kk * 32, half);
        v16bf b0 = *(const v16bf*)(vrow0 + kk * 32);
        v16bf b1 = *(const v16bf*)(vrow1 + kk * 32);
        acc00 = wmma_bf16(a0, b0, acc00);
        acc01 = wmma_bf16(a0, b1, acc01);
        acc10 = wmma_bf16(a1, b0, acc10);
        acc11 = wmma_bf16(a1, b1, acc11);
    }
#pragma unroll
    for (int r = 0; r < 8; ++r) {
        int M = r + half * 8;
        O[((size_t)b * SEQ + i0 + M) * DK + d0 + nm]           = acc00[r];
        O[((size_t)b * SEQ + i0 + M) * DK + d0 + 16 + nm]      = acc01[r];
        O[((size_t)b * SEQ + i0 + 16 + M) * DK + d0 + nm]      = acc10[r];
        O[((size_t)b * SEQ + i0 + 16 + M) * DK + d0 + 16 + nm] = acc11[r];
    }
}

// ---------------------------------------------------------------------------
extern "C" void kernel_launch(void* const* d_in, const int* in_sizes, int n_in,
                              void* d_out, int out_size, void* d_ws, size_t ws_size,
                              hipStream_t stream) {
    const float* x  = (const float*)d_in[0];
    const float* Wq = (const float*)d_in[1];
    const float* bq = (const float*)d_in[2];
    const float* Wk = (const float*)d_in[3];
    const float* bk = (const float*)d_in[4];
    const float* Wv = (const float*)d_in[5];
    const float* bv = (const float*)d_in[6];
    float* out = (float*)d_out;

    char* ws = (char*)d_ws;
    const size_t M = (size_t)BATCH * SEQ;            // 16384 rows
    size_t off = 0;
    bf16* X16 = (bf16*)(ws + off); off += M * DK * sizeof(bf16);          // 32 MB
    bf16* WTq = (bf16*)(ws + off); off += (size_t)DK * DK * sizeof(bf16); //  2 MB
    bf16* WTk = (bf16*)(ws + off); off += (size_t)DK * DK * sizeof(bf16);
    bf16* WTv = (bf16*)(ws + off); off += (size_t)DK * DK * sizeof(bf16);
    bf16* Q16 = (bf16*)(ws + off); off += M * DK * sizeof(bf16);          // 32 MB
    bf16* K16 = (bf16*)(ws + off); off += M * DK * sizeof(bf16);          // 32 MB
    bf16* Vt16= (bf16*)(ws + off); off += M * DK * sizeof(bf16);          // 32 MB
    float* SC = (float*)(ws + off); off += (size_t)BATCH * SEQ * SEQ * sizeof(float); // 128 MB

    // 1) precision prep
    size_t nx = M * DK;
    cvt_f32_bf16<<<dim3((unsigned)((nx + 255) / 256)), 256, 0, stream>>>(x, X16, nx);
    transpose_w<<<dim3(DK / 32, DK / 32), 256, 0, stream>>>(Wq, WTq);
    transpose_w<<<dim3(DK / 32, DK / 32), 256, 0, stream>>>(Wk, WTk);
    transpose_w<<<dim3(DK / 32, DK / 32), 256, 0, stream>>>(Wv, WTv);

    // 2) Q/K/V projections (8 waves per block, one 32x32 tile per wave)
    qkv_gemm<<<dim3(DK / 32, (unsigned)(M / 256), 3), 256, 0, stream>>>(
        X16, WTq, WTk, WTv, bq, bk, bv, Q16, K16, Vt16);

    // 3) causal score tiles (32x32 blocks, lower triangle)
    scores_kernel<<<dim3(SEQ / 32, SEQ / 256, BATCH), 256, 0, stream>>>(Q16, K16, SC);

    // 4) row softmax -> bf16 probabilities in place
    softmax_rows<<<dim3((unsigned)M), 256, 0, stream>>>(SC);

    // 5) O = P @ V (32x32 tiles)
    av_gemm<<<dim3(DK / 32, SEQ / 256, BATCH), 256, 0, stream>>>(SC, Vt16, out);
}
